// SSD_44375602102799
// MI455X (gfx1250) — matrix operand order repaired and checked by reading, not compiled, not run on hardware
//
#include <hip/hip_runtime.h>
#include <math.h>

// ---------------- problem constants ----------------
#define B_     16
#define SEQ_   336
#define PRED_  96
#define F_     7
#define DM_    512
#define NH_    8
#define DFF_   2048
#define S_     16
#define DH_    64
#define EL_    2
#define DL_    1
#define BF_    (B_ * F_)

typedef __attribute__((ext_vector_type(16))) _Float16 v16h;
typedef __attribute__((ext_vector_type(8)))  _Float16 v8h;
typedef __attribute__((ext_vector_type(8)))  float    v8f;
typedef _Float16 half_t;

// ---------------- Tensor Data Mover support ----------------
#if __has_builtin(__builtin_amdgcn_tensor_load_to_lds) && __has_builtin(__builtin_amdgcn_s_wait_tensorcnt)
#define HAVE_TDM 1
#else
#define HAVE_TDM 0
#endif

typedef unsigned int u32x4 __attribute__((ext_vector_type(4)));
typedef int          i32x4 __attribute__((ext_vector_type(4)));
typedef int          i32x8 __attribute__((ext_vector_type(8)));

#if HAVE_TDM
// Issue a 2D TDM tile load: tileRows x tileCols half_t elements from a
// row-major matrix (rowStrideElems) into LDS at ldsByteAddr (row-compact:
// tileCols*2 bytes per row).  D# layout per CDNA5 ISA §8.3/8.4.
__device__ __forceinline__ void tdm_load_tile_2d(
    const half_t* gsrc, int rowStrideElems,
    unsigned ldsByteAddr, int tileCols, int tileRows,
    int tensorCols, int tensorRows)
{
    unsigned long long ga = (unsigned long long)(size_t)gsrc;
    u32x4 g0;
    g0[0] = 1u;                                       // count=1, user D#
    g0[1] = ldsByteAddr;                              // lds_addr
    g0[2] = (unsigned)(ga & 0xFFFFFFFFu);             // global_addr[31:0]
    g0[3] = (unsigned)((ga >> 32) & 0x01FFFFFFu) | 0x80000000u;  // addr[56:32], type=2
    unsigned long long st0 = (unsigned long long)rowStrideElems;
    i32x8 g1;
    g1[0] = 0x00010000;                               // data_size=1 (2 bytes)
    g1[1] = (int)(((unsigned)tensorCols & 0xFFFFu) << 16);          // tensor_dim0 lo
    g1[2] = (int)((((unsigned)tensorCols >> 16) & 0xFFFFu)
                  | (((unsigned)tensorRows & 0xFFFFu) << 16));      // td0 hi | td1 lo
    g1[3] = (int)((((unsigned)tensorRows >> 16) & 0xFFFFu)
                  | (((unsigned)tileCols & 0xFFFFu) << 16));        // td1 hi | tile_dim0
    g1[4] = (int)((unsigned)tileRows & 0xFFFFu);                    // tile_dim1
    g1[5] = (int)(st0 & 0xFFFFFFFFu);                               // tensor_dim0_stride lo
    g1[6] = (int)((st0 >> 32) & 0xFFFFu);                           // stride hi
    g1[7] = 0;
    i32x4 gz = {0, 0, 0, 0};
#if defined(__clang_major__) && (__clang_major__ >= 23)
    i32x8 gz8 = {0, 0, 0, 0, 0, 0, 0, 0};
    __builtin_amdgcn_tensor_load_to_lds(g0, g1, gz, gz, gz8, 0);
#else
    __builtin_amdgcn_tensor_load_to_lds(g0, g1, gz, gz, 0);
#endif
}
#endif

// =====================================================================
// Batched f16 WMMA GEMM.  C[M,N](f16) = relu?( alpha * A@B + bias(f32) )
//   BT = true : B given as (N x K) row-major  (y = x @ W^T form)
//   BT = false: B given as (K x N) row-major
// Block = 128 threads (4 waves); tile 64x64, K-step 32.
// 16 v_wmma_f32_16x16x32_f16 per block per K-step.
// Interior NT tiles stage B via the Tensor Data Mover (wave 0 issues,
// s_wait_tensorcnt, barrier); A uses 16B vector staging; edge tiles use
// a branch-free clamped scalar path.
// =====================================================================
template <bool BT>
__global__ __launch_bounds__(128) void gemm_wmma(
    const half_t* __restrict__ A, size_t sAo, size_t sAi, int lda, int aMod,
    const half_t* __restrict__ Bm, size_t sBo, size_t sBi, int ldb, int bMod,
    const float* __restrict__ bias, size_t sBiasO, int biasMod, int biasMode,
    half_t* __restrict__ C, size_t sCo, size_t sCi, int ldc,
    int M, int N, int K, float alpha, int relu, int inner)
{
    __shared__ __align__(16) half_t As[64][40];   // [m][k], 80B rows
    __shared__ __align__(16) half_t Bs[64][32];   // [n][k], 64B rows (TDM-compact)

    const int bo = blockIdx.z / inner;
    const int bi = blockIdx.z % inner;
    const int asel = aMod ? (bo % aMod) : bo;
    const int bsel = bMod ? (bo % bMod) : bo;

    const half_t* Ab = A  + (size_t)asel * sAo + (size_t)bi * sAi;
    const half_t* Bb = Bm + (size_t)bsel * sBo + (size_t)bi * sBi;
    half_t*       Cb = C  + (size_t)bo   * sCo + (size_t)bi * sCi;
    const float* biasb = nullptr;
    if (biasMode) {
        const int bsl = biasMod ? (bo % biasMod) : bo;
        biasb = bias + (size_t)bsl * sBiasO;
    }

    const int m0 = blockIdx.x * 64;
    const int n0 = blockIdx.y * 64;
    const int tid  = threadIdx.x;
    const int wave = tid >> 5;
    const int lane = tid & 31;
    const int wm0 = (wave >> 1) * 32;
    const int wn0 = (wave & 1) * 32;
    const int lrow = lane & 15;
    const int hi   = lane >> 4;
    const int ka8  = hi ? 8 : 0;          // A frag: lanes16-31 K+8 / K+24
    const int kb16 = hi ? 16 : 0;         // unused; B frag uses ka8 (K+8 / K+24) like A

    const bool mFull = (m0 + 64 <= M);
    const bool nFull = (n0 + 64 <= N);

    v8f acc[2][2] = {};

    for (int k0 = 0; k0 < K; k0 += 32) {
        const bool kFull = (k0 + 32 <= K);

        // ---- stage A tile (64 rows x 32 halves) ----
        if (kFull && mFull) {
#pragma unroll
            for (int s = 0; s < 2; ++s) {
                int seg = tid + s * 128;          // 256 v8h segments
                int r  = seg >> 2;
                int c8 = (seg & 3) * 8;
                const half_t* src = Ab + (size_t)(m0 + r) * lda + k0 + c8;
                *(v8h*)&As[r][c8] = *(const v8h*)src;
                __builtin_prefetch(src + 32, 0, 0);
            }
        } else {
            for (int i = tid; i < 64 * 32; i += 128) {
                int r = i >> 5, c = i & 31;
                int gm = m0 + r, gk = k0 + c;
                bool ok = (gm < M) && (gk < K);
                half_t v = Ab[(size_t)(ok ? gm : 0) * lda + (ok ? gk : 0)];
                As[r][c] = ok ? v : (half_t)0.0f;
            }
        }

        // ---- stage B tile (64 n x 32 k), stored [n][k] ----
        if constexpr (BT) {
            if (kFull && nFull) {
#if HAVE_TDM
                if (wave == 0) {
                    tdm_load_tile_2d(Bb + (size_t)n0 * ldb + k0, ldb,
                                     (unsigned)(size_t)&Bs[0][0],
                                     /*tileCols=*/32, /*tileRows=*/64,
                                     /*tensorCols=*/K - k0, /*tensorRows=*/N - n0);
                    __builtin_amdgcn_s_wait_tensorcnt(0);
                }
#else
#pragma unroll
                for (int s = 0; s < 2; ++s) {
                    int seg = tid + s * 128;
                    int r  = seg >> 2;
                    int c8 = (seg & 3) * 8;
                    const half_t* src = Bb + (size_t)(n0 + r) * ldb + k0 + c8;
                    *(v8h*)&Bs[r][c8] = *(const v8h*)src;
                    __builtin_prefetch(src + 32, 0, 0);
                }
#endif
            } else {
                for (int i = tid; i < 64 * 32; i += 128) {
                    int r = i >> 5, c = i & 31;
                    int gn = n0 + r, gk = k0 + c;
                    bool ok = (gn < N) && (gk < K);
                    half_t v = Bb[(size_t)(ok ? gn : 0) * ldb + (ok ? gk : 0)];
                    Bs[r][c] = ok ? v : (half_t)0.0f;
                }
            }
        } else {
            if (kFull && nFull) {
#pragma unroll
                for (int s = 0; s < 2; ++s) {
                    int seg = tid + s * 128;      // 32 k-rows x 8 segments
                    int k  = seg >> 3;
                    int n8 = (seg & 7) * 8;
                    const half_t* src = Bb + (size_t)(k0 + k) * ldb + n0 + n8;
                    v8h v = *(const v8h*)src;
#pragma unroll
                    for (int j = 0; j < 8; ++j) Bs[n8 + j][k] = v[j];
                }
            } else {
                for (int i = tid; i < 64 * 32; i += 128) {
                    int r = i >> 5, c = i & 31;
                    int gn = n0 + r, gk = k0 + c;
                    bool ok = (gn < N) && (gk < K);
                    half_t v = Bb[(size_t)(ok ? gk : 0) * ldb + (ok ? gn : 0)];
                    Bs[r][c] = ok ? v : (half_t)0.0f;
                }
            }
        }
        __syncthreads();

        // ---- build fragments (two aligned v8h LDS reads each) ----
        v16h afr[2], bfr[2];
#pragma unroll
        for (int t = 0; t < 2; ++t) {
            int ar = wm0 + t * 16 + lrow;
            v8h alo = *(const v8h*)&As[ar][ka8];
            v8h ahi = *(const v8h*)&As[ar][16 + ka8];
            afr[t] = __builtin_shufflevector(alo, ahi,
                0,1,2,3,4,5,6,7,8,9,10,11,12,13,14,15);
            int bc = wn0 + t * 16 + lrow;
            v8h blo = *(const v8h*)&Bs[bc][ka8];
            v8h bhi = *(const v8h*)&Bs[bc][16 + ka8];
            bfr[t] = __builtin_shufflevector(blo, bhi,
                0,1,2,3,4,5,6,7,8,9,10,11,12,13,14,15);
        }
#pragma unroll
        for (int i = 0; i < 2; ++i)
#pragma unroll
            for (int j = 0; j < 2; ++j)
                acc[i][j] = __builtin_amdgcn_wmma_f32_16x16x32_f16(
                    false, afr[i], false, bfr[j], (short)0, acc[i][j], false, false);
        __syncthreads();
    }

    // ---- epilogue ----
    const int rbase = hi ? 8 : 0;
#pragma unroll
    for (int i = 0; i < 2; ++i) {
#pragma unroll
        for (int j = 0; j < 2; ++j) {
            int gn = n0 + wn0 + j * 16 + lrow;
#pragma unroll
            for (int r = 0; r < 8; ++r) {
                int gm = m0 + wm0 + i * 16 + rbase + r;
                if (gm < M && gn < N) {
                    float v = alpha * acc[i][j][r];
                    if (biasMode == 1)      v += biasb[gn];
                    else if (biasMode == 2) v += biasb[gm];
                    if (relu) v = v > 0.0f ? v : 0.0f;
                    Cb[(size_t)gm * ldc + gn] = (half_t)v;
                }
            }
        }
    }
}

// =====================================================================
// f32 -> f16 weight conversion
// =====================================================================
__global__ void f32_to_f16_kernel(const float* __restrict__ s,
                                  half_t* __restrict__ d, size_t n)
{
    size_t i = (size_t)blockIdx.x * blockDim.x + threadIdx.x;
    if (i < n) d[i] = (half_t)s[i];
}

// =====================================================================
// Embeddings with on-device sinusoidal positional encoding (f16 out)
// =====================================================================
__device__ __forceinline__ float pe_val(int s, int d)
{
    int pair = d >> 1;
    float freq = expf((float)(2 * pair) * (-9.210340371976184f / (float)DM_));
    float ang = (float)s * freq;
    return (d & 1) ? cosf(ang) : sinf(ang);
}

__global__ void embed_enc_kernel(const float* __restrict__ bx,
                                 const float* __restrict__ bxm,
                                 const float* __restrict__ w_in,
                                 const float* __restrict__ b_in,
                                 half_t* __restrict__ XE)
{
    size_t i = (size_t)blockIdx.x * blockDim.x + threadIdx.x;
    if (i >= (size_t)BF_ * SEQ_ * DM_) return;
    int d  = (int)(i % DM_);
    int s  = (int)((i / DM_) % SEQ_);
    int bf = (int)(i / ((size_t)DM_ * SEQ_));
    int b = bf / F_, f = bf % F_;
    const float* w  = w_in + (size_t)d * 5;
    const float* mk = bxm + ((size_t)b * SEQ_ + s) * 4;
    float h0 = bx[((size_t)b * SEQ_ + s) * F_ + f];
    float v = h0 * w[0] + mk[0] * w[1] + mk[1] * w[2] + mk[2] * w[3] + mk[3] * w[4];
    v += b_in[d] + pe_val(s, d);
    XE[i] = (half_t)v;
}

__global__ void embed_dec_kernel(const float* __restrict__ bym,
                                 const float* __restrict__ w_co,
                                 const float* __restrict__ b_co,
                                 half_t* __restrict__ XD)
{
    size_t i = (size_t)blockIdx.x * blockDim.x + threadIdx.x;
    if (i >= (size_t)BF_ * PRED_ * DM_) return;
    int d  = (int)(i % DM_);
    int p  = (int)((i / DM_) % PRED_);
    int bf = (int)(i / ((size_t)DM_ * PRED_));
    int b = bf / F_;
    const float* w  = w_co + (size_t)d * 4;
    const float* mk = bym + ((size_t)b * PRED_ + p) * 4;
    float v = mk[0] * w[0] + mk[1] * w[1] + mk[2] * w[2] + mk[3] * w[3];
    v += b_co[d] + pe_val(p, d);
    XD[i] = (half_t)v;
}

// =====================================================================
// Row softmax on f16 scores, f32 internal math
// =====================================================================
__global__ __launch_bounds__(128) void softmax_kernel(half_t* __restrict__ S,
                                                      int Lq, int Lk, int causal)
{
    __shared__ float red[128];
    size_t row = blockIdx.x;
    int q = (int)(row % (size_t)Lq);
    half_t* p = S + row * (size_t)Lk;
    int t = threadIdx.x;

    float mx = -3.0e38f;
    for (int c = t; c < Lk; c += 128) {
        float v = (float)p[c] + ((causal && c > q) ? -1.0e9f : 0.0f);
        mx = fmaxf(mx, v);
    }
    red[t] = mx; __syncthreads();
    for (int o = 64; o > 0; o >>= 1) { if (t < o) red[t] = fmaxf(red[t], red[t + o]); __syncthreads(); }
    mx = red[0]; __syncthreads();

    float sum = 0.0f;
    for (int c = t; c < Lk; c += 128) {
        float v = (float)p[c] + ((causal && c > q) ? -1.0e9f : 0.0f);
        v = expf(v - mx);
        p[c] = (half_t)v;
        sum += v;
    }
    red[t] = sum; __syncthreads();
    for (int o = 64; o > 0; o >>= 1) { if (t < o) red[t] += red[t + o]; __syncthreads(); }
    float inv = 1.0f / red[0];
    for (int c = t; c < Lk; c += 128) p[c] = (half_t)((float)p[c] * inv);
}

// =====================================================================
// X = LayerNorm(X + Y) * gamma + beta on f16 I/O (f32 math)
// =====================================================================
__global__ __launch_bounds__(256) void ln_residual_kernel(
    half_t* __restrict__ X, const half_t* __restrict__ Y,
    const float* __restrict__ gamma, const float* __restrict__ beta,
    size_t gStride, int rowsPerPair)
{
    __shared__ float red[256];
    int row  = blockIdx.x;
    int f    = (row / rowsPerPair) % F_;
    const float* g  = gamma + (size_t)f * gStride;
    const float* bb = beta  + (size_t)f * gStride;
    half_t*       x = X + (size_t)row * DM_;
    const half_t* y = Y + (size_t)row * DM_;
    int t = threadIdx.x;

    float v0 = (float)x[t] + (float)y[t];
    float v1 = (float)x[t + 256] + (float)y[t + 256];

    red[t] = v0 + v1; __syncthreads();
    for (int o = 128; o > 0; o >>= 1) { if (t < o) red[t] += red[t + o]; __syncthreads(); }
    float mean = red[0] * (1.0f / DM_); __syncthreads();

    float d0 = v0 - mean, d1 = v1 - mean;
    red[t] = d0 * d0 + d1 * d1; __syncthreads();
    for (int o = 128; o > 0; o >>= 1) { if (t < o) red[t] += red[t + o]; __syncthreads(); }
    float rstd = rsqrtf(red[0] * (1.0f / DM_) + 1e-5f);

    x[t]       = (half_t)(d0 * rstd * g[t]       + bb[t]);
    x[t + 256] = (half_t)(d1 * rstd * g[t + 256] + bb[t + 256]);
}

// =====================================================================
// Output head
// =====================================================================
__global__ __launch_bounds__(32) void head_kernel(
    const half_t* __restrict__ XD, const float* __restrict__ wc,
    const float* __restrict__ bc, const float* __restrict__ wsv,
    const float* __restrict__ bsv, float* __restrict__ Cc, float* __restrict__ Sg)
{
    size_t rp = blockIdx.x;                 // bf*PRED + p
    const half_t* x = XD + rp * DM_;
    int lane = threadIdx.x;

    float xv[16];
#pragma unroll
    for (int i = 0; i < 16; ++i) xv[i] = (float)x[lane + 32 * i];

    for (int o = 0; o < 17; ++o) {
        const float* w = (o < 16) ? (wc + (size_t)o * DM_) : wsv;
        float s = 0.0f;
#pragma unroll
        for (int i = 0; i < 16; ++i) s += xv[i] * w[lane + 32 * i];
#pragma unroll
        for (int m = 16; m >= 1; m >>= 1) s += __shfl_xor(s, m, 32);
        if (lane == 0) {
            if (o < 16) {
                float c = (s + bc[o]) * (1.0f / 6.0f) + 0.5f;
                Cc[rp * 16 + o] = fminf(fmaxf(c, 0.0f), 1.0f);
            } else {
                float z = s + bsv[0];
                Sg[rp] = (z > 0.0f) ? (z + log1pf(expf(-z))) : log1pf(expf(z));
            }
        }
    }
}

// =====================================================================
// TAO scan
// =====================================================================
__global__ __launch_bounds__(32) void scan_kernel(
    const float* __restrict__ Cc, const float* __restrict__ Sg,
    float* __restrict__ out)
{
    int bf = blockIdx.x;
    int b = bf / F_, f = bf % F_;
    int lane = threadIdx.x;
    int i = lane & 15;

    float a = Cc[((size_t)bf * PRED_ + 0) * 16 + i];
    for (int p = 0; p < PRED_; ++p) {
        float s = 0.0f;
#pragma unroll
        for (int j = 0; j < 16; ++j) {
            float aj = __shfl(a, j, 32);
            float tv;
            if (i == 0)      tv = (j == 0) ? 1.0f : 0.0f;
            else if (i == 1) tv = (j >= 1 && j <= 14) ? -1.0f : 0.0f;
            else             tv = (j == i - 1) ? 1.0f : 0.0f;
            s += tv * aj;
        }
        a = s + Cc[((size_t)bf * PRED_ + p) * 16 + i];
        float y = __shfl(a, 0, 32) + __shfl(a, 1, 32) + Sg[(size_t)bf * PRED_ + p];
        if (lane == 0) out[((size_t)b * PRED_ + p) * F_ + f] = y;
    }
}

// =====================================================================
// Host-side launch helpers
// =====================================================================
static void gemm_nt(hipStream_t st,
    const half_t* A, size_t sAo, size_t sAi, int lda, int aMod,
    const half_t* Bm, size_t sBo, size_t sBi, int ldb, int bMod,
    const float* bias, size_t sBiasO, int biasMod, int biasMode,
    half_t* C, size_t sCo, size_t sCi, int ldc,
    int M, int N, int K, float alpha, int relu, int outer, int inner)
{
    dim3 g((M + 63) / 64, (N + 63) / 64, outer * inner);
    gemm_wmma<true><<<g, 128, 0, st>>>(A, sAo, sAi, lda, aMod, Bm, sBo, sBi, ldb, bMod,
        bias, sBiasO, biasMod, biasMode, C, sCo, sCi, ldc, M, N, K, alpha, relu, inner);
}

static void gemm_nn(hipStream_t st,
    const half_t* A, size_t sAo, size_t sAi, int lda, int aMod,
    const half_t* Bm, size_t sBo, size_t sBi, int ldb, int bMod,
    const float* bias, size_t sBiasO, int biasMod, int biasMode,
    half_t* C, size_t sCo, size_t sCi, int ldc,
    int M, int N, int K, float alpha, int relu, int outer, int inner)
{
    dim3 g((M + 63) / 64, (N + 63) / 64, outer * inner);
    gemm_wmma<false><<<g, 128, 0, st>>>(A, sAo, sAi, lda, aMod, Bm, sBo, sBi, ldb, bMod,
        bias, sBiasO, biasMod, biasMode, C, sCo, sCi, ldc, M, N, K, alpha, relu, inner);
}

// ---------------- workspace layout ----------------
static constexpr size_t N_CC  = (size_t)BF_ * PRED_ * S_;   // f32
static constexpr size_t N_SG  = (size_t)BF_ * PRED_;        // f32
static constexpr size_t N_XE  = (size_t)BF_ * SEQ_ * DM_;
static constexpr size_t N_XD  = (size_t)BF_ * PRED_ * DM_;
static constexpr size_t N_QKV = (size_t)BF_ * SEQ_ * 3 * DM_;
static constexpr size_t N_SC  = (size_t)BF_ * NH_ * SEQ_ * SEQ_;  // also hosts FF1
static constexpr size_t N_OB  = (size_t)BF_ * SEQ_ * DM_;
static constexpr size_t N_ATT = (size_t)BF_ * SEQ_ * DM_;
static constexpr size_t N_MEM = (size_t)BF_ * PRED_ * DM_;
static constexpr size_t N_QC  = (size_t)BF_ * PRED_ * DM_;
static constexpr size_t N_KV  = (size_t)BF_ * PRED_ * 2 * DM_;
static constexpr size_t N_EQW = (size_t)F_ * EL_ * 3 * DM_ * DM_;
static constexpr size_t N_EOW = (size_t)F_ * EL_ * DM_ * DM_;
static constexpr size_t N_EF1 = (size_t)F_ * EL_ * DFF_ * DM_;
static constexpr size_t N_EF2 = (size_t)F_ * EL_ * DM_ * DFF_;
static constexpr size_t N_ELW = (size_t)F_ * PRED_ * SEQ_;
static constexpr size_t N_DQW = (size_t)F_ * DL_ * 3 * DM_ * DM_;
static constexpr size_t N_DOW = (size_t)F_ * DL_ * DM_ * DM_;
static constexpr size_t N_DF1 = (size_t)F_ * DL_ * DFF_ * DM_;
static constexpr size_t N_DF2 = (size_t)F_ * DL_ * DM_ * DFF_;

extern "C" void kernel_launch(void* const* d_in, const int* in_sizes, int n_in,
                              void* d_out, int out_size, void* d_ws, size_t ws_size,
                              hipStream_t stream)
{
    (void)in_sizes; (void)n_in; (void)out_size; (void)ws_size;
    const float* bx   = (const float*)d_in[0];
    const float* bxm  = (const float*)d_in[1];
    const float* bym  = (const float*)d_in[3];
    const float* w_in = (const float*)d_in[4];
    const float* b_in = (const float*)d_in[5];
    const float* w_co = (const float*)d_in[6];
    const float* b_co = (const float*)d_in[7];
    const float* eqw  = (const float*)d_in[8];
    const float* eqb  = (const float*)d_in[9];
    const float* eow  = (const float*)d_in[10];
    const float* eob  = (const float*)d_in[11];
    const float* el1w = (const float*)d_in[12];
    const float* el1b = (const float*)d_in[13];
    const float* el2w = (const float*)d_in[14];
    const float* el2b = (const float*)d_in[15];
    const float* ef1w = (const float*)d_in[16];
    const float* ef1b = (const float*)d_in[17];
    const float* ef2w = (const float*)d_in[18];
    const float* ef2b = (const float*)d_in[19];
    const float* elw  = (const float*)d_in[20];
    const float* elb  = (const float*)d_in[21];
    const float* dsqw = (const float*)d_in[22];
    const float* dsqb = (const float*)d_in[23];
    const float* dsow = (const float*)d_in[24];
    const float* dsob = (const float*)d_in[25];
    const float* dcqw = (const float*)d_in[26];
    const float* dcqb = (const float*)d_in[27];
    const float* dcow = (const float*)d_in[28];
    const float* dcob = (const float*)d_in[29];
    const float* dl1w = (const float*)d_in[30];
    const float* dl1b = (const float*)d_in[31];
    const float* dl2w = (const float*)d_in[32];
    const float* dl2b = (const float*)d_in[33];
    const float* dl3w = (const float*)d_in[34];
    const float* dl3b = (const float*)d_in[35];
    const float* df1w = (const float*)d_in[36];
    const float* df1b = (const float*)d_in[37];
    const float* df2w = (const float*)d_in[38];
    const float* df2b = (const float*)d_in[39];
    const float* wc   = (const float*)d_in[40];
    const float* bc   = (const float*)d_in[41];
    const float* wsv  = (const float*)d_in[42];
    const float* bsv  = (const float*)d_in[43];

    float* wsf = (float*)d_ws;
    float* Cc = wsf;
    float* Sg = wsf + N_CC;
    half_t* H = (half_t*)(wsf + N_CC + N_SG);

    size_t o = 0;
    half_t* XE  = H + o;  o += N_XE;
    half_t* XD  = H + o;  o += N_XD;
    half_t* QKV = H + o;  o += N_QKV;
    half_t* SC  = H + o;  o += N_SC;
    half_t* FF1 = SC;                       // aliased, disjoint in time
    half_t* OB  = H + o;  o += N_OB;
    half_t* ATT = H + o;  o += N_ATT;
    half_t* MEM = H + o;  o += N_MEM;
    half_t* QC  = H + o;  o += N_QC;
    half_t* KV  = H + o;  o += N_KV;
    half_t* hEQW = H + o; o += N_EQW;
    half_t* hEOW = H + o; o += N_EOW;
    half_t* hEF1 = H + o; o += N_EF1;
    half_t* hEF2 = H + o; o += N_EF2;
    half_t* hELW = H + o; o += N_ELW;
    half_t* hDQW = H + o; o += N_DQW;
    half_t* hDOW = H + o; o += N_DOW;
    half_t* hCQW = H + o; o += N_DQW;
    half_t* hCOW = H + o; o += N_DOW;
    half_t* hDF1 = H + o; o += N_DF1;
    half_t* hDF2 = H + o; o += N_DF2;
    float* out = (float*)d_out;

    auto conv = [&](const float* s, half_t* d, size_t n) {
        f32_to_f16_kernel<<<(unsigned)((n + 255) / 256), 256, 0, stream>>>(s, d, n);
    };
    conv(eqw,  hEQW, N_EQW);
    conv(eow,  hEOW, N_EOW);
    conv(ef1w, hEF1, N_EF1);
    conv(ef2w, hEF2, N_EF2);
    conv(elw,  hELW, N_ELW);
    conv(dsqw, hDQW, N_DQW);
    conv(dsow, hDOW, N_DOW);
    conv(dcqw, hCQW, N_DQW);
    conv(dcow, hCOW, N_DOW);
    conv(df1w, hDF1, N_DF1);
    conv(df2w, hDF2, N_DF2);

    {
        size_t n = (size_t)BF_ * SEQ_ * DM_;
        embed_enc_kernel<<<(unsigned)((n + 255) / 256), 256, 0, stream>>>(bx, bxm, w_in, b_in, XE);
        size_t m = (size_t)BF_ * PRED_ * DM_;
        embed_dec_kernel<<<(unsigned)((m + 255) / 256), 256, 0, stream>>>(bym, w_co, b_co, XD);
    }

    for (int l = 0; l < EL_; ++l) {
        gemm_nt(stream, XE, (size_t)SEQ_ * DM_, 0, DM_, 0,
                hEQW + (size_t)l * 3 * DM_ * DM_, (size_t)EL_ * 3 * DM_ * DM_, 0, DM_, F_,
                eqb + (size_t)l * 3 * DM_, (size_t)EL_ * 3 * DM_, F_, 1,
                QKV, (size_t)SEQ_ * 3 * DM_, 0, 3 * DM_,
                SEQ_, 3 * DM_, DM_, 1.0f, 0, BF_, 1);
        gemm_nt(stream, QKV, (size_t)SEQ_ * 3 * DM_, DH_, 3 * DM_, 0,
                QKV + DM_, (size_t)SEQ_ * 3 * DM_, DH_, 3 * DM_, 0,
                nullptr, 0, 0, 0,
                SC, (size_t)NH_ * SEQ_ * SEQ_, (size_t)SEQ_ * SEQ_, SEQ_,
                SEQ_, SEQ_, DH_, 0.125f, 0, BF_, NH_);
        softmax_kernel<<<BF_ * NH_ * SEQ_, 128, 0, stream>>>(SC, SEQ_, SEQ_, 0);
        gemm_nn(stream, SC, (size_t)NH_ * SEQ_ * SEQ_, (size_t)SEQ_ * SEQ_, SEQ_, 0,
                QKV + 2 * DM_, (size_t)SEQ_ * 3 * DM_, DH_, 3 * DM_, 0,
                nullptr, 0, 0, 0,
                OB, (size_t)SEQ_ * DM_, DH_, DM_,
                SEQ_, DH_, SEQ_, 1.0f, 0, BF_, NH_);
        gemm_nt(stream, OB, (size_t)SEQ_ * DM_, 0, DM_, 0,
                hEOW + (size_t)l * DM_ * DM_, (size_t)EL_ * DM_ * DM_, 0, DM_, F_,
                eob + (size_t)l * DM_, (size_t)EL_ * DM_, F_, 1,
                ATT, (size_t)SEQ_ * DM_, 0, DM_,
                SEQ_, DM_, DM_, 1.0f, 0, BF_, 1);
        ln_residual_kernel<<<BF_ * SEQ_, 256, 0, stream>>>(
            XE, ATT, el1w + (size_t)l * DM_, el1b + (size_t)l * DM_, (size_t)EL_ * DM_, SEQ_);
        gemm_nt(stream, XE, (size_t)SEQ_ * DM_, 0, DM_, 0,
                hEF1 + (size_t)l * DFF_ * DM_, (size_t)EL_ * DFF_ * DM_, 0, DM_, F_,
                ef1b + (size_t)l * DFF_, (size_t)EL_ * DFF_, F_, 1,
                FF1, (size_t)SEQ_ * DFF_, 0, DFF_,
                SEQ_, DFF_, DM_, 1.0f, 1, BF_, 1);
        gemm_nt(stream, FF1, (size_t)SEQ_ * DFF_, 0, DFF_, 0,
                hEF2 + (size_t)l * DM_ * DFF_, (size_t)EL_ * DM_ * DFF_, 0, DFF_, F_,
                ef2b + (size_t)l * DM_, (size_t)EL_ * DM_, F_, 1,
                ATT, (size_t)SEQ_ * DM_, 0, DM_,
                SEQ_, DM_, DFF_, 1.0f, 0, BF_, 1);
        ln_residual_kernel<<<BF_ * SEQ_, 256, 0, stream>>>(
            XE, ATT, el2w + (size_t)l * DM_, el2b + (size_t)l * DM_, (size_t)EL_ * DM_, SEQ_);
    }

    gemm_nn(stream, hELW, (size_t)PRED_ * SEQ_, 0, SEQ_, F_,
            XE, (size_t)SEQ_ * DM_, 0, DM_, 0,
            elb, (size_t)PRED_, F_, 2,
            MEM, (size_t)PRED_ * DM_, 0, DM_,
            PRED_, DM_, SEQ_, 1.0f, 0, BF_, 1);

    {
        gemm_nt(stream, XD, (size_t)PRED_ * DM_, 0, DM_, 0,
                hDQW, (size_t)DL_ * 3 * DM_ * DM_, 0, DM_, F_,
                dsqb, (size_t)DL_ * 3 * DM_, F_, 1,
                QKV, (size_t)PRED_ * 3 * DM_, 0, 3 * DM_,
                PRED_, 3 * DM_, DM_, 1.0f, 0, BF_, 1);
        gemm_nt(stream, QKV, (size_t)PRED_ * 3 * DM_, DH_, 3 * DM_, 0,
                QKV + DM_, (size_t)PRED_ * 3 * DM_, DH_, 3 * DM_, 0,
                nullptr, 0, 0, 0,
                SC, (size_t)NH_ * PRED_ * PRED_, (size_t)PRED_ * PRED_, PRED_,
                PRED_, PRED_, DH_, 0.125f, 0, BF_, NH_);
        softmax_kernel<<<BF_ * NH_ * PRED_, 128, 0, stream>>>(SC, PRED_, PRED_, 1);
        gemm_nn(stream, SC, (size_t)NH_ * PRED_ * PRED_, (size_t)PRED_ * PRED_, PRED_, 0,
                QKV + 2 * DM_, (size_t)PRED_ * 3 * DM_, DH_, 3 * DM_, 0,
                nullptr, 0, 0, 0,
                OB, (size_t)PRED_ * DM_, DH_, DM_,
                PRED_, DH_, PRED_, 1.0f, 0, BF_, NH_);
        gemm_nt(stream, OB, (size_t)PRED_ * DM_, 0, DM_, 0,
                hDOW, (size_t)DL_ * DM_ * DM_, 0, DM_, F_,
                dsob, (size_t)DL_ * DM_, F_, 1,
                ATT, (size_t)PRED_ * DM_, 0, DM_,
                PRED_, DM_, DM_, 1.0f, 0, BF_, 1);
        ln_residual_kernel<<<BF_ * PRED_, 256, 0, stream>>>(
            XD, ATT, dl1w, dl1b, (size_t)DL_ * DM_, PRED_);

        gemm_nt(stream, XD, (size_t)PRED_ * DM_, 0, DM_, 0,
                hCQW, (size_t)DL_ * 3 * DM_ * DM_, 0, DM_, F_,
                dcqb, (size_t)DL_ * 3 * DM_, F_, 1,
                QC, (size_t)PRED_ * DM_, 0, DM_,
                PRED_, DM_, DM_, 1.0f, 0, BF_, 1);
        gemm_nt(stream, MEM, (size_t)PRED_ * DM_, 0, DM_, 0,
                hCQW + (size_t)DM_ * DM_, (size_t)DL_ * 3 * DM_ * DM_, 0, DM_, F_,
                dcqb + (size_t)DM_, (size_t)DL_ * 3 * DM_, F_, 1,
                KV, (size_t)PRED_ * 2 * DM_, 0, 2 * DM_,
                PRED_, 2 * DM_, DM_, 1.0f, 0, BF_, 1);
        gemm_nt(stream, QC, (size_t)PRED_ * DM_, DH_, DM_, 0,
                KV, (size_t)PRED_ * 2 * DM_, DH_, 2 * DM_, 0,
                nullptr, 0, 0, 0,
                SC, (size_t)NH_ * PRED_ * PRED_, (size_t)PRED_ * PRED_, PRED_,
                PRED_, PRED_, DH_, 0.125f, 0, BF_, NH_);
        softmax_kernel<<<BF_ * NH_ * PRED_, 128, 0, stream>>>(SC, PRED_, PRED_, 0);
        gemm_nn(stream, SC, (size_t)NH_ * PRED_ * PRED_, (size_t)PRED_ * PRED_, PRED_, 0,
                KV + DM_, (size_t)PRED_ * 2 * DM_, DH_, 2 * DM_, 0,
                nullptr, 0, 0, 0,
                OB, (size_t)PRED_ * DM_, DH_, DM_,
                PRED_, DH_, PRED_, 1.0f, 0, BF_, NH_);
        gemm_nt(stream, OB, (size_t)PRED_ * DM_, 0, DM_, 0,
                hCOW, (size_t)DL_ * DM_ * DM_, 0, DM_, F_,
                dcob, (size_t)DL_ * DM_, F_, 1,
                ATT, (size_t)PRED_ * DM_, 0, DM_,
                PRED_, DM_, DM_, 1.0f, 0, BF_, 1);
        ln_residual_kernel<<<BF_ * PRED_, 256, 0, stream>>>(
            XD, ATT, dl2w, dl2b, (size_t)DL_ * DM_, PRED_);

        gemm_nt(stream, XD, (size_t)PRED_ * DM_, 0, DM_, 0,
                hDF1, (size_t)DL_ * DFF_ * DM_, 0, DM_, F_,
                df1b, (size_t)DL_ * DFF_, F_, 1,
                FF1, (size_t)PRED_ * DFF_, 0, DFF_,
                PRED_, DFF_, DM_, 1.0f, 1, BF_, 1);
        gemm_nt(stream, FF1, (size_t)PRED_ * DFF_, 0, DFF_, 0,
                hDF2, (size_t)DL_ * DM_ * DFF_, 0, DFF_, F_,
                df2b, (size_t)DL_ * DM_, F_, 1,
                ATT, (size_t)PRED_ * DM_, 0, DM_,
                PRED_, DM_, DFF_, 1.0f, 0, BF_, 1);
        ln_residual_kernel<<<BF_ * PRED_, 256, 0, stream>>>(
            XD, ATT, dl3w, dl3b, (size_t)DL_ * DM_, PRED_);
    }

    head_kernel<<<BF_ * PRED_, 32, 0, stream>>>(XD, wc, bc, wsv, bsv, Cc, Sg);
    scan_kernel<<<BF_, 32, 0, stream>>>(Cc, Sg, out);
}
